// DecoderAttentionSingle_41918880808996
// MI455X (gfx1250) — compile-verified, matrix operand order checked by reading, not run on hardware
//
#include <hip/hip_runtime.h>

typedef _Float16 f16;
typedef __attribute__((ext_vector_type(16))) _Float16 v16h;
typedef __attribute__((ext_vector_type(8)))  _Float16 v8h;
typedef __attribute__((ext_vector_type(8)))  float    v8f;
typedef __attribute__((ext_vector_type(4)))  float    v4f;

static constexpr int kB   = 8;
static constexpr int kENC = 64;
static constexpr int kDEC = 128;
static constexpr int kH   = 128;
static constexpr int kW   = 128;
static constexpr int kHW  = kH * kW;          // 16384 = 2^14
static constexpr int kNPIX = kB * kHW;        // 131072
static constexpr int KP128 = 136;             // padded LDS pixel stride (halves): 16B-aligned, bank-skewed
static constexpr int KP64  = 72;

__device__ __forceinline__ v8f zero8() {
  v8f z;
#pragma unroll
  for (int i = 0; i < 8; ++i) z[i] = 0.0f;
  return z;
}

__device__ __forceinline__ v8f wmma_f16(v16h a, v16h b, v8f c) {
  // v_wmma_f32_16x16x32_f16  D = A*B + C
  return __builtin_amdgcn_wmma_f32_16x16x32_f16(false, a, false, b, (short)0, c, false, false);
}

// Load a 16(row) x 32(K) f16 fragment (LDS or global), rows `stride` halves apart.
// ISA 16-bit A/B layout: lane L holds row (L&15), K halves kb..kb+7 and kb+16..kb+23,
// kb = (L>>4)*8.  Two 16B loads per fragment.
__device__ __forceinline__ v16h load_frag(const f16* p, int stride) {
  const int lane = threadIdx.x & 31;
  const int r    = lane & 15;
  const int kb   = (lane >> 4) << 3;
  const f16* q = p + r * stride + kb;
  v8h lo = *(const v8h*)(q);
  v8h hi = *(const v8h*)(q + 16);
  v16h a;
#pragma unroll
  for (int i = 0; i < 8; ++i) { a[i] = lo[i]; a[i + 8] = hi[i]; }
  return a;
}

// Wave-local LDS store->load ordering (tiles are private per wave; no workgroup barrier).
__device__ __forceinline__ void wave_lds_fence() {
  __builtin_amdgcn_wave_barrier();
  asm volatile("s_wait_dscnt 0" ::: "memory");
  __builtin_amdgcn_wave_barrier();
}

__device__ __forceinline__ float fast_tanh(float x) {
  x = fminf(10.0f, fmaxf(-10.0f, x));
  float e = __expf(2.0f * x);
  return (e - 1.0f) / (e + 1.0f);
}

__device__ __forceinline__ void load32h(const f16* p, float* dst) {
#pragma unroll
  for (int j = 0; j < 4; ++j) {
    v8h t = *(const v8h*)(p + 8 * j);
#pragma unroll
    for (int i = 0; i < 8; ++i) dst[8 * j + i] = (float)t[i];
  }
}

// ---------------------------------------------------------------------------
// Kernel 0: one-time f32 -> f16 transposed weight staging into workspace.
// Layouts: [n][k] with k contiguous so B-fragments are two global b128 loads.
// Total ~188 KB -> permanently L2-resident, broadcast to all blocks.
// ---------------------------------------------------------------------------
__global__ __launch_bounds__(256)
void prep_weights_kernel(const float* __restrict__ W_dec, const float* __restrict__ W_enc,
                         const float* __restrict__ W_attn, const float* __restrict__ conv_w,
                         f16* __restrict__ wWdT, f16* __restrict__ wWeT,
                         f16* __restrict__ wWaT, f16* __restrict__ wCvT)
{
  const int i0 = blockIdx.x * 256 + threadIdx.x;
  const int stride = gridDim.x * 256;
  for (int i = i0; i < kDEC * kENC; i += stride) {        // W_dec [k][n] -> [n][128]
    int k = i >> 6, n = i & 63;
    wWdT[n * kDEC + k] = (f16)W_dec[i];
  }
  for (int i = i0; i < kENC * kENC; i += stride) {        // W_enc [k][n] -> [n][64]
    int k = i >> 6, n = i & 63;
    wWeT[n * kENC + k] = (f16)W_enc[i];
  }
  for (int i = i0; i < kDEC * kENC; i += stride) {        // W_attn [k][n] -> [n][128]
    int k = i >> 6, n = i & 63;
    wWaT[n * kDEC + k] = (f16)W_attn[i];
  }
  for (int i = i0; i < 9 * kDEC * kENC; i += stride) {    // conv_w [t][k][n] -> [t][n][128]
    int n = i & 63, k = (i >> 6) & 127, t = i >> 13;
    wCvT[((size_t)t * kENC + n) * kDEC + k] = (f16)conv_w[i];
  }
}

// ---------------------------------------------------------------------------
// Kernel 1: q = dec@W_dec+b_dec, k = enc@W_enc+b_enc (f16, pixel-major [p][64]),
//           plus f16 pixel-major copy of raw enc for the attention gather.
// 256 threads = 8 waves, 16 pixels/wave. No block barriers; float4 staging.
// ---------------------------------------------------------------------------
__global__ __launch_bounds__(256)
void proj_qk_kernel(const float* __restrict__ enc, const float* __restrict__ dec,
                    const float* __restrict__ b_enc, const float* __restrict__ b_dec,
                    const f16* __restrict__ wWdT, const f16* __restrict__ wWeT,
                    f16* __restrict__ wsq, f16* __restrict__ wsk, f16* __restrict__ wse)
{
  __shared__ __align__(16) f16 sA[8][16 * KP128];  // dec tiles, one per wave
  __shared__ __align__(16) f16 sE[8][16 * KP64];   // enc tiles, one per wave

  const int tid = threadIdx.x, lane = tid & 31, wave = tid >> 5;
  __builtin_prefetch(wWdT, 0, 3);
  __builtin_prefetch(wWeT, 0, 3);

  const int tile_p = blockIdx.x * 128 + wave * 16;
  const int b   = tile_p >> 14;
  const int hw0 = tile_p & (kHW - 1);

  // NCHW -> [pixel][channel] f16 tiles (float4 loads: tiles never cross a row)
  f16* A = sA[wave];
  for (int i = lane; i < 16 * kDEC / 4; i += 32) {
    int c = i >> 2, m4 = (i & 3) << 2;
    v4f d = *(const v4f*)(dec + (((size_t)b * kDEC + c) << 14) + hw0 + m4);
#pragma unroll
    for (int u = 0; u < 4; ++u) A[(m4 + u) * KP128 + c] = (f16)d[u];
  }
  f16* E = sE[wave];
  for (int i = lane; i < 16 * kENC / 4; i += 32) {
    int c = i >> 2, m4 = (i & 3) << 2;
    v4f d = *(const v4f*)(enc + (((size_t)b * kENC + c) << 14) + hw0 + m4);
#pragma unroll
    for (int u = 0; u < 4; ++u) {
      f16 v = (f16)d[u];
      E[(m4 + u) * KP64 + c] = v;
      wse[((size_t)(tile_p + m4 + u) << 6) + c] = v;
    }
  }
  wave_lds_fence();

  const int col = lane & 15, m0 = (lane >> 4) << 3;

  // q: K=128 (4 k-steps), N=64 (4 n-tiles); B fragments streamed from L2-resident ws
  v8f accq[4];
#pragma unroll
  for (int nt = 0; nt < 4; ++nt) accq[nt] = zero8();
#pragma unroll
  for (int ks = 0; ks < 4; ++ks) {
    v16h a = load_frag(A + ks * 32, KP128);
#pragma unroll
    for (int nt = 0; nt < 4; ++nt)
      accq[nt] = wmma_f16(a, load_frag(wWdT + (nt * 16) * kDEC + ks * 32, kDEC), accq[nt]);
  }
#pragma unroll
  for (int nt = 0; nt < 4; ++nt) {
    int n = nt * 16 + col;
    float bias = b_dec[n];
#pragma unroll
    for (int r = 0; r < 8; ++r)
      wsq[((size_t)(tile_p + m0 + r) << 6) + n] = (f16)(accq[nt][r] + bias);
  }

  // k: K=64 (2 k-steps)
  v8f acck[4];
#pragma unroll
  for (int nt = 0; nt < 4; ++nt) acck[nt] = zero8();
#pragma unroll
  for (int ks = 0; ks < 2; ++ks) {
    v16h a = load_frag(E + ks * 32, KP64);
#pragma unroll
    for (int nt = 0; nt < 4; ++nt)
      acck[nt] = wmma_f16(a, load_frag(wWeT + (nt * 16) * kENC + ks * 32, kENC), acck[nt]);
  }
#pragma unroll
  for (int nt = 0; nt < 4; ++nt) {
    int n = nt * 16 + col;
    float bias = b_enc[n];
#pragma unroll
    for (int r = 0; r < 8; ++r)
      wsk[((size_t)(tile_p + m0 + r) << 6) + n] = (f16)(acck[nt][r] + bias);
  }
}

// ---------------------------------------------------------------------------
// Kernel 2: vals = conv3x3(dec) + conv_b.  Per wave: stage a 3x18-pixel x 128ch
// halo tile ONCE, then 9 taps = shifted fragment bases into the same LDS tile.
// 128 threads = 4 waves, 16 pixels/wave. Weights streamed from ws (L2).
// ---------------------------------------------------------------------------
__global__ __launch_bounds__(128)
void conv_vals_kernel(const float* __restrict__ dec, const f16* __restrict__ wCvT,
                      const float* __restrict__ conv_b, f16* __restrict__ wsv)
{
  __shared__ __align__(16) f16 sH[4][3 * 18 * KP128]; // [row 0..2][px 0..17][c]

  const int tid = threadIdx.x, lane = tid & 31, wave = tid >> 5;
  __builtin_prefetch(wCvT, 0, 3);

  const int tile_p = blockIdx.x * 64 + wave * 16;
  const int b   = tile_p >> 14;
  const int hw0 = tile_p & (kHW - 1);
  const int h0  = hw0 >> 7;
  const int w0  = hw0 & (kW - 1);

  // stage halo: rows h0-1..h0+1, pixels w0-1..w0+16, zero-filled at borders
  f16* H = sH[wave];
  for (int i = lane; i < 3 * kDEC * 18; i += 32) {
    int px  = i % 18;
    int t2  = i / 18;
    int c   = t2 & 127;
    int row = t2 >> 7;
    int hh = h0 + row - 1, ww = w0 + px - 1;
    float v = 0.0f;
    if (((unsigned)hh < (unsigned)kH) && ((unsigned)ww < (unsigned)kW))
      v = dec[(((size_t)b * kDEC + c) << 14) + (hh << 7) + ww];
    H[(row * 18 + px) * KP128 + c] = (f16)v;
  }
  wave_lds_fence();

  v8f acc[4];
#pragma unroll
  for (int nt = 0; nt < 4; ++nt) acc[nt] = zero8();

#pragma unroll
  for (int t = 0; t < 9; ++t) {
    const int dy = t / 3, dx = t - (t / 3) * 3;            // 0..2
    const f16* Ab = H + (dy * 18 + dx) * KP128;            // pixel m -> halo px m+dx
    const f16* Wb = wCvT + ((size_t)t * kENC) * kDEC;
#pragma unroll
    for (int ks = 0; ks < 4; ++ks) {
      v16h a = load_frag(Ab + ks * 32, KP128);
#pragma unroll
      for (int nt = 0; nt < 4; ++nt)
        acc[nt] = wmma_f16(a, load_frag(Wb + (nt * 16) * kDEC + ks * 32, kDEC), acc[nt]);
    }
  }

  const int col = lane & 15, m0 = (lane >> 4) << 3;
#pragma unroll
  for (int nt = 0; nt < 4; ++nt) {
    int n = nt * 16 + col;
    float bias = conv_b[n];
#pragma unroll
    for (int r = 0; r < 8; ++r)
      wsv[((size_t)(tile_p + m0 + r) << 6) + n] = (f16)(acc[nt][r] + bias);
  }
}

// ---------------------------------------------------------------------------
// Kernel 3: neighborhood tanh-attention + softmax + gather, then
//           out = LeakyReLU([vals ‖ attn] @ W_attn + b_attn), stored NCHW.
// 256 threads = 8 waves; 2 lanes per pixel (32 channels each). No block barriers.
// ---------------------------------------------------------------------------
__global__ __launch_bounds__(256)
void attn_out_kernel(const f16* __restrict__ wsq, const f16* __restrict__ wsk,
                     const f16* __restrict__ wse, const f16* __restrict__ wsv,
                     const float* __restrict__ W_agg, const float* __restrict__ b_agg,
                     const f16* __restrict__ wWaT, const float* __restrict__ b_attn,
                     float* __restrict__ out)
{
  __shared__ __align__(16) f16 sCat[8][16 * KP128];

  const int tid = threadIdx.x, lane = tid & 31, wave = tid >> 5;
  __builtin_prefetch(wWaT, 0, 3);

  const int tile_p = blockIdx.x * 128 + wave * 16;
  const int b   = tile_p >> 14;
  const int hw0 = tile_p & (kHW - 1);
  const int h0  = hw0 >> 7;
  const int w0  = hw0 & (kW - 1);

  const int j   = lane & 15;        // pixel within tile
  const int ch0 = (lane >> 4) * 32; // channel chunk (0 or 32)
  const int p   = tile_p + j;
  const int w   = w0 + j;

  // per-lane W_agg chunk (L1-broadcast)
  float wg[32];
#pragma unroll
  for (int u = 0; u < 8; ++u) {
    v4f t = *(const v4f*)(W_agg + ch0 + u * 4);
#pragma unroll
    for (int i = 0; i < 4; ++i) wg[u * 4 + i] = t[i];
  }
  const float bagg = b_agg[0];

  // ---- additive attention scores over 3x3 neighborhood ----
  float qv[32];
  load32h(wsq + ((size_t)p << 6) + ch0, qv);

  float sc[9];
#pragma unroll
  for (int n = 0; n < 9; ++n) {
    const int dy = n / 3 - 1, dx = n - (n / 3) * 3 - 1;
    const int hh = h0 + dy, ww = w + dx;
    const bool ok = ((unsigned)hh < (unsigned)kH) && ((unsigned)ww < (unsigned)kW);
    float s = 0.0f;
    if (ok) {
      const size_t np = ((size_t)b << 14) + (hh << 7) + ww;
      float kv[32];
      load32h(wsk + (np << 6) + ch0, kv);
#pragma unroll
      for (int c = 0; c < 32; ++c)
        s += fast_tanh(qv[c] + kv[c]) * wg[c];
    }
    s += __shfl_xor(s, 16, 32);          // combine the two half-dots of this pixel
    sc[n] = ok ? (s + bagg) : -1.0e30f;  // masked fill
  }

  // ---- softmax over 9 ----
  float mx = sc[0];
#pragma unroll
  for (int n = 1; n < 9; ++n) mx = fmaxf(mx, sc[n]);
  float wgt[9], ssum = 0.0f;
#pragma unroll
  for (int n = 0; n < 9; ++n) { wgt[n] = __expf(sc[n] - mx); ssum += wgt[n]; }
  const float inv = 1.0f / ssum;

  // ---- cat tile: [vals(64) | attn_vec(64)] per pixel ----
  f16* cat = sCat[wave];
  {
    float vv[32];
    load32h(wsv + ((size_t)p << 6) + ch0, vv);
#pragma unroll
    for (int c = 0; c < 32; ++c) cat[j * KP128 + ch0 + c] = (f16)vv[c];
  }
  float av[32];
#pragma unroll
  for (int c = 0; c < 32; ++c) av[c] = 0.0f;
#pragma unroll
  for (int n = 0; n < 9; ++n) {
    const int dy = n / 3 - 1, dx = n - (n / 3) * 3 - 1;
    const int hh = h0 + dy, ww = w + dx;
    if (((unsigned)hh < (unsigned)kH) && ((unsigned)ww < (unsigned)kW)) {
      const float wn = wgt[n] * inv;
      float ev[32];
      load32h(wse + ((((size_t)b << 14) + (hh << 7) + ww) << 6) + ch0, ev);
#pragma unroll
      for (int c = 0; c < 32; ++c) av[c] += wn * ev[c];
    }
  }
#pragma unroll
  for (int c = 0; c < 32; ++c) cat[j * KP128 + 64 + ch0 + c] = (f16)av[c];

  wave_lds_fence();

  // ---- final GEMM: cat(16x128) @ W_attn(128x64) ----
  v8f acc[4];
#pragma unroll
  for (int nt = 0; nt < 4; ++nt) acc[nt] = zero8();
#pragma unroll
  for (int ks = 0; ks < 4; ++ks) {
    v16h a = load_frag(cat + ks * 32, KP128);
#pragma unroll
    for (int nt = 0; nt < 4; ++nt)
      acc[nt] = wmma_f16(a, load_frag(wWaT + (nt * 16) * kDEC + ks * 32, kDEC), acc[nt]);
  }

  // ---- bias + LeakyReLU(0.2) + coalesced NCHW store ----
  const int col = lane & 15, m0 = (lane >> 4) << 3;
#pragma unroll
  for (int nt = 0; nt < 4; ++nt) {
    const int n = nt * 16 + col;
    const float bias = b_attn[n];
    float o[8];
#pragma unroll
    for (int r = 0; r < 8; ++r) {
      float v = acc[nt][r] + bias;
      o[r] = (v >= 0.0f) ? v : 0.2f * v;
    }
    float* base = out + (((size_t)b * kENC + n) << 14) + hw0 + m0;
    v4f lo = { o[0], o[1], o[2], o[3] };
    v4f hi = { o[4], o[5], o[6], o[7] };
    *(v4f*)(base)     = lo;
    *(v4f*)(base + 4) = hi;
  }
}

// ---------------------------------------------------------------------------
extern "C" void kernel_launch(void* const* d_in, const int* in_sizes, int n_in,
                              void* d_out, int out_size, void* d_ws, size_t ws_size,
                              hipStream_t stream) {
  (void)in_sizes; (void)n_in; (void)out_size; (void)ws_size;

  const float* enc    = (const float*)d_in[0];
  const float* dec    = (const float*)d_in[1];
  const float* W_enc  = (const float*)d_in[2];
  const float* b_enc  = (const float*)d_in[3];
  const float* W_dec  = (const float*)d_in[4];
  const float* b_dec  = (const float*)d_in[5];
  const float* W_agg  = (const float*)d_in[6];
  const float* b_agg  = (const float*)d_in[7];
  const float* W_attn = (const float*)d_in[8];
  const float* b_attn = (const float*)d_in[9];
  const float* conv_w = (const float*)d_in[10];
  const float* conv_b = (const float*)d_in[11];
  float* out = (float*)d_out;

  // workspace: four f16 [NPIX][64] activations, then f16 transposed weights
  f16* wsq = (f16*)d_ws;
  f16* wsk = wsq + (size_t)kNPIX * kENC;
  f16* wse = wsk + (size_t)kNPIX * kENC;
  f16* wsv = wse + (size_t)kNPIX * kENC;
  f16* wWdT = wsv  + (size_t)kNPIX * kENC;   // [64][128]
  f16* wWeT = wWdT + kENC * kDEC;            // [64][64]
  f16* wWaT = wWeT + kENC * kENC;            // [64][128]
  f16* wCvT = wWaT + kENC * kDEC;            // [9][64][128]

  prep_weights_kernel<<<dim3(64), dim3(256), 0, stream>>>(
      W_dec, W_enc, W_attn, conv_w, wWdT, wWeT, wWaT, wCvT);

  proj_qk_kernel<<<dim3(kNPIX / 128), dim3(256), 0, stream>>>(
      enc, dec, b_enc, b_dec, wWdT, wWeT, wsq, wsk, wse);

  conv_vals_kernel<<<dim3(kNPIX / 64), dim3(128), 0, stream>>>(
      dec, wCvT, conv_b, wsv);

  attn_out_kernel<<<dim3(kNPIX / 128), dim3(256), 0, stream>>>(
      wsq, wsk, wse, wsv, W_agg, b_agg, wWaT, b_attn, out);
}
